// SurrogateModel_18562848653973
// MI455X (gfx1250) — matrix operand order repaired and checked by reading, not compiled, not run on hardware
//
#include <hip/hip_runtime.h>
#include <hip/hip_bf16.h>
#include <math.h>

typedef __bf16 bf16_t;
typedef __attribute__((ext_vector_type(16))) __bf16 v16bf;
typedef __attribute__((ext_vector_type(8)))  float  v8f;

#define F 256   // F_IN == F_HID == F_OUT

// ---- monotone encoding so float max can use integer atomicMax ----
__device__ __forceinline__ unsigned enc_f32(float f) {
    unsigned u = __float_as_uint(f);
    return (u & 0x80000000u) ? ~u : (u | 0x80000000u);
}
__device__ __forceinline__ float dec_f32(unsigned k) {
    return (k & 0x80000000u) ? __uint_as_float(k & 0x7fffffffu)
                             : __uint_as_float(~k);
}
__device__ __forceinline__ float sigm(float x) { return 1.0f / (1.0f + __expf(-x)); }

// ---------------- init kernels ----------------
__global__ void k_init_nodes(unsigned* __restrict__ mu, float* __restrict__ z, int nn) {
    int n = blockIdx.x * blockDim.x + threadIdx.x;
    if (n < nn) { mu[n] = 0u; z[n] = 0.0f; }   // enc 0 == -NaN sentinel (identity for max)
}

// out rows start at bias b2 (gc2 "+ b"): grid = nn blocks of 256
__global__ void k_init_out(float* __restrict__ out, const float* __restrict__ b2) {
    out[(size_t)blockIdx.x * F + threadIdx.x] = b2[threadIdx.x];
}

// ---------------- h = bf16(x @ W2) via WMMA ----------------
// block = 256 threads = 8 waves; wave w: m-tile = blockIdx.x*8+w,
// TWO n-tiles (blockIdx.y*32, +16): two independent accumulator chains so
// consecutive v_wmma ops have no D->C RAW hazard, and A tiles are reused 2x.
__global__ void __launch_bounds__(256)
k_gemm_xW2(const float* __restrict__ x, const float* __restrict__ W2,
           bf16_t* __restrict__ h, int nn) {
    const int lane  = threadIdx.x & 31;
    const int wave  = threadIdx.x >> 5;
    const int mtiles = (nn + 15) >> 4;
    const int mtile  = blockIdx.x * 8 + wave;
    if (mtile >= mtiles) return;                 // wave-uniform: EXEC stays all-1s
    const int m0   = mtile << 4;
    const int n0   = blockIdx.y << 5;            // 32 output columns per wave
    const int half = lane >> 4;                  // lane group 0-15 / 16-31
    int mA = m0 + (lane & 15);
    if (mA >= nn) mA = nn - 1;                   // clamp (nn%16==0 in practice)

    v8f c0 = {}, c1 = {};
    #pragma unroll
    for (int kk = 0; kk < 8; ++kk) {             // K = 256 in steps of 32
        const int k0 = kk << 5;
        // A tile 16x32: lane holds row M=lane&15; K = (e>>3)*16 + half*8 + (e&7)
        const float* ap = x + (size_t)mA * F + k0 + half * 8;
        float4 a0 = *(const float4*)(ap);
        float4 a1 = *(const float4*)(ap + 4);
        float4 a2 = *(const float4*)(ap + 16);
        float4 a3 = *(const float4*)(ap + 20);
        v16bf a;
        a[0]=(bf16_t)a0.x; a[1]=(bf16_t)a0.y; a[2]=(bf16_t)a0.z; a[3]=(bf16_t)a0.w;
        a[4]=(bf16_t)a1.x; a[5]=(bf16_t)a1.y; a[6]=(bf16_t)a1.z; a[7]=(bf16_t)a1.w;
        a[8]=(bf16_t)a2.x; a[9]=(bf16_t)a2.y; a[10]=(bf16_t)a2.z; a[11]=(bf16_t)a2.w;
        a[12]=(bf16_t)a3.x; a[13]=(bf16_t)a3.y; a[14]=(bf16_t)a3.z; a[15]=(bf16_t)a3.w;
        // B tiles 32x16: lane holds column N=n0+(lane&15) (bB: +16); K = (e>>3)*16 + half*8 + (e&7)
        const float* bp = W2 + (size_t)(k0 + half * 8) * F + n0 + (lane & 15);
        v16bf bA, bB;
        #pragma unroll
        for (int e = 0; e < 16; ++e) {
            const float* q = bp + (size_t)((e >> 3) * 16 + (e & 7)) * F;
            bA[e] = (bf16_t)q[0];
            bB[e] = (bf16_t)q[16];
        }

        c0 = __builtin_amdgcn_wmma_f32_16x16x32_bf16(false, a, false, bA,
                                                     (short)0, c0, false, false);
        c1 = __builtin_amdgcn_wmma_f32_16x16x32_bf16(false, a, false, bB,
                                                     (short)0, c1, false, false);
    }
    // C/D layout: lane l, vgpr r -> (M = r + 8*half, N = l&15)
    const int nc = n0 + (lane & 15);
    #pragma unroll
    for (int r = 0; r < 8; ++r) {
        int m = m0 + r + half * 8;
        if (m < nn) {
            h[(size_t)m * F + nc]      = (bf16_t)c0[r];
            h[(size_t)m * F + nc + 16] = (bf16_t)c1[r];
        }
    }
}

// ---------------- per-node attention scores ----------------
__global__ void __launch_bounds__(256)
k_scores(const bf16_t* __restrict__ h, const float* __restrict__ asrc,
         const float* __restrict__ adst, float* __restrict__ ssrc,
         float* __restrict__ sdst, int nn) {
    __shared__ float as[F], ad[F];
    as[threadIdx.x] = asrc[threadIdx.x];
    ad[threadIdx.x] = adst[threadIdx.x];
    __syncthreads();
    int n = blockIdx.x * blockDim.x + threadIdx.x;
    if (n >= nn) return;
    const bf16_t* hp = h + (size_t)n * F;
    float s0 = 0.f, s1 = 0.f;
    #pragma unroll 4
    for (int k = 0; k < F; ++k) {
        float v = (float)hp[k];
        s0 += v * as[k]; s1 += v * ad[k];
    }
    ssrc[n] = s0; sdst[n] = s1;
}

// ---------------- edge logits + segment max ----------------
__global__ void k_logits(const int* __restrict__ ei, const float* __restrict__ ssrc,
                         const float* __restrict__ sdst, float* __restrict__ lbuf,
                         unsigned* __restrict__ mu, int E, int nn) {
    int i = blockIdx.x * blockDim.x + threadIdx.x;
    if (i >= E + nn) return;
    int s, d;
    if (i < E) { s = ei[i]; d = ei[E + i]; } else { s = d = i - E; }   // self loops
    float v = ssrc[s] + sdst[d];
    v = v > 0.f ? v : 0.2f * v;                                       // leaky_relu(0.2)
    lbuf[i] = v;
    atomicMax(&mu[d], enc_f32(v));
}

// ---------------- exp + segment sum ----------------
__global__ void k_expz(const int* __restrict__ ei, float* __restrict__ lbuf,
                       const unsigned* __restrict__ mu, float* __restrict__ z,
                       int E, int nn) {
    int i = blockIdx.x * blockDim.x + threadIdx.x;
    if (i >= E + nn) return;
    int d = (i < E) ? ei[E + i] : (i - E);
    float e = __expf(lbuf[i] - dec_f32(mu[d]));
    lbuf[i] = e;
    atomicAdd(&z[d], e);
}

// ---------------- message scatter: out[dst] += alpha * h[src] ----------------
__global__ void __launch_bounds__(256)
k_scatter(const int* __restrict__ ei, const float* __restrict__ ebuf,
          const float* __restrict__ z, const bf16_t* __restrict__ h,
          float* __restrict__ out, int E, int nn) {
    const int ET = E + nn;
    for (int i = blockIdx.x; i < ET; i += gridDim.x) {
        int s, d;
        if (i < E) { s = ei[i]; d = ei[E + i]; } else { s = d = i - E; }
        float alpha = ebuf[i] / z[d];
        float v = alpha * (float)h[(size_t)s * F + threadIdx.x];
        atomicAdd(&out[(size_t)d * F + threadIdx.x], v);
    }
}

// ---------------- fused gx + LSTM scan + fc head (1 WGP, 32 wave32) ----------------
__global__ void __launch_bounds__(1024)
k_lstm(const float* __restrict__ h2, const float* __restrict__ W_ih,
       const float* __restrict__ W_hh, const float* __restrict__ b_ih,
       const float* __restrict__ b_hh, const float* __restrict__ Wfc,
       const float* __restrict__ bfc, float* __restrict__ out, int nn) {
    __shared__ __align__(16) float hs_sh[F];
    __shared__ __align__(16) float xr[F];
    __shared__ float g_sh[4 * F];
    const int j = threadIdx.x;                 // gate-row 0..1023 (order i,f,g,o)
    const float bsum = b_ih[j] + b_hh[j];
    float cs = 0.f;
    if (j < F) hs_sh[j] = 0.f;
    const float4* wih = (const float4*)(W_ih + (size_t)j * F);   // L2-resident (1 MB)
    const float4* whh = (const float4*)(W_hh + (size_t)j * F);   // L2-resident (1 MB)
    __syncthreads();
    for (int t = 0; t < nn; ++t) {
        if (j < F) {
            xr[j] = h2[(size_t)t * F + j];
            // pull next step's row toward the WGP while this step computes
            // (global_prefetch_b8: no LOADcnt cost, off the critical path)
            if (t + 1 < nn) __builtin_prefetch(&h2[(size_t)(t + 1) * F + j], 0, 1);
        }
        __syncthreads();
        const float4* xv = (const float4*)xr;
        const float4* hv = (const float4*)hs_sh;
        float acc = bsum;
        #pragma unroll 8
        for (int q = 0; q < F / 4; ++q) {      // 512 FMAs: W_ih[j]*x_t + W_hh[j]*h
            float4 wa = wih[q], wb = whh[q];
            float4 xa = xv[q],  hb = hv[q];
            acc += wa.x * xa.x + wa.y * xa.y + wa.z * xa.z + wa.w * xa.w;
            acc += wb.x * hb.x + wb.y * hb.y + wb.z * hb.z + wb.w * hb.w;
        }
        g_sh[j] = acc;
        __syncthreads();
        if (j < F) {
            float iv = sigm(g_sh[j]);
            float fv = sigm(g_sh[F + j]);
            float gv = tanhf(g_sh[2 * F + j]);
            float ov = sigm(g_sh[3 * F + j]);
            cs = fv * cs + iv * gv;
            hs_sh[j] = ov * tanhf(cs);
        }
        __syncthreads();
    }
    // head: relu(c_n) @ Wfc.T + bfc
    if (j < F) g_sh[j] = fmaxf(cs, 0.f) * Wfc[j];
    __syncthreads();
    if (j == 0) {
        float s = bfc[0];
        for (int k = 0; k < F; ++k) s += g_sh[k];
        out[0] = s;
    }
}

extern "C" void kernel_launch(void* const* d_in, const int* in_sizes, int n_in,
                              void* d_out, int out_size, void* d_ws, size_t ws_size,
                              hipStream_t stream) {
    (void)n_in; (void)out_size; (void)ws_size;
    const float* x    = (const float*)d_in[0];
    const int*   ei   = (const int*)d_in[1];
    // d_in[2] edge_attr unused (edge_dim=None); d_in[3..6] gc1 params: dead code, elided
    const float* W2   = (const float*)d_in[7];
    const float* a2s  = (const float*)d_in[8];
    const float* a2d  = (const float*)d_in[9];
    const float* b2   = (const float*)d_in[10];
    const float* W_ih = (const float*)d_in[11];
    const float* W_hh = (const float*)d_in[12];
    const float* b_ih = (const float*)d_in[13];
    const float* b_hh = (const float*)d_in[14];
    const float* Wfc  = (const float*)d_in[15];
    const float* bfc  = (const float*)d_in[16];

    const int NN = in_sizes[0] / F;        // 50000 nodes
    const int E  = in_sizes[1] / 2;        // 800000 edges
    const int ET = E + NN;                 // + self loops

    // workspace carve-out (~81 MB total)
    char* w = (char*)d_ws;
    auto carve = [&](size_t bytes) -> void* {
        void* p = (void*)w;
        w += (bytes + 255) & ~(size_t)255;
        return p;
    };
    bf16_t*   h_bf = (bf16_t*)  carve((size_t)NN * F * sizeof(bf16_t)); // 25.6 MB
    float*    h2   = (float*)   carve((size_t)NN * F * sizeof(float));  // 51.2 MB
    float*    ssrc = (float*)   carve((size_t)NN * sizeof(float));
    float*    sdst = (float*)   carve((size_t)NN * sizeof(float));
    unsigned* mu   = (unsigned*)carve((size_t)NN * sizeof(unsigned));
    float*    z    = (float*)   carve((size_t)NN * sizeof(float));
    float*    ebuf = (float*)   carve((size_t)ET * sizeof(float));      // 3.4 MB

    const int nb_nodes = (NN + 255) / 256;
    const int nb_edges = (ET + 255) / 256;
    const int mtiles   = (NN + 15) / 16;

    k_init_nodes<<<nb_nodes, 256, 0, stream>>>(mu, z, NN);
    k_init_out  <<<NN, 256, 0, stream>>>(h2, b2);
    {
        dim3 g((mtiles + 7) / 8, F / 32);   // 8 n-tile pairs
        k_gemm_xW2<<<g, 256, 0, stream>>>(x, W2, h_bf, NN);
    }
    k_scores <<<nb_nodes, 256, 0, stream>>>(h_bf, a2s, a2d, ssrc, sdst, NN);
    k_logits <<<nb_edges, 256, 0, stream>>>(ei, ssrc, sdst, ebuf, mu, E, NN);
    k_expz   <<<nb_edges, 256, 0, stream>>>(ei, ebuf, mu, z, E, NN);
    k_scatter<<<4096, 256, 0, stream>>>(ei, ebuf, z, h_bf, h2, E, NN);
    k_lstm   <<<1, 1024, 0, stream>>>(h2, W_ih, W_hh, b_ih, b_hh, Wfc, bfc,
                                      (float*)d_out, NN);
}
